// WindowAttention_51367808860813
// MI455X (gfx1250) — matrix operand order repaired and checked by reading, not compiled, not run on hardware
//
#include <hip/hip_runtime.h>

typedef __attribute__((ext_vector_type(16))) __bf16 v16bf;
typedef __attribute__((ext_vector_type(8)))  float  v8f;

#define WMMA_BF16(a, b, c) \
    __builtin_amdgcn_wmma_f32_16x16x32_bf16(false, (a), false, (b), (short)0, (c), false, false)

constexpr int kB = 32, kNW = 64, kL = 49, kD = 192, kPROJ = 256, kHEADS = 8, kHD = 32;
constexpr int kBNW = kB * kNW;   // 2048
constexpr int kM   = kBNW * kL;  // 100352 rows (multiple of 16)
constexpr int kLP  = 64;         // L padded to 4 tiles of 16

// ---------------------------------------------------------------------------
// Kernel 0: convert weights to bf16 and transpose to [N][K] for contiguous
// WMMA B-fragment loads.
// ---------------------------------------------------------------------------
__global__ void pack_weights(const float* __restrict__ Wq, const float* __restrict__ Wk,
                             const float* __restrict__ Wv, const float* __restrict__ Ww,
                             __bf16* __restrict__ WqT, __bf16* __restrict__ WkT,
                             __bf16* __restrict__ WvT, __bf16* __restrict__ WwT) {
    int idx = blockIdx.x * 256 + threadIdx.x;
    const int per = kD * kPROJ;  // 49152
    if (idx < 3 * per) {
        int which = idx / per, r = idx % per;
        int n = r / kD, k = r % kD;  // out [PROJ][D]
        const float* W = which == 0 ? Wq : which == 1 ? Wk : Wv;
        __bf16* O = which == 0 ? WqT : which == 1 ? WkT : WvT;
        O[n * kD + k] = (__bf16)W[k * kPROJ + n];
    } else if (idx < 4 * per) {
        int r = idx - 3 * per;
        int n = r / kPROJ, k = r % kPROJ;  // out [D][PROJ]
        WwT[n * kPROJ + k] = (__bf16)Ww[k * kD + n];
    }
}

// ---------------------------------------------------------------------------
// Kernel 1: fused Q/K/V projection.  grid=(M/16, 1, 3), block=128 (4 waves).
// Each wave keeps ONE A fragment and drives it into 4 N-tiles (4 accumulators),
// so the block covers all 256 output columns and the streaming A tensor is
// read once per block instead of once per N-tile.
// ---------------------------------------------------------------------------
__global__ void qkv_proj(const float* __restrict__ q, const float* __restrict__ kx,
                         const float* __restrict__ v,
                         const __bf16* __restrict__ WqT, const __bf16* __restrict__ WkT,
                         const __bf16* __restrict__ WvT,
                         const float* __restrict__ bq, const float* __restrict__ bk,
                         __bf16* __restrict__ qh, __bf16* __restrict__ kh,
                         __bf16* __restrict__ vh) {
    int sel = blockIdx.z;
    const float*  X    = sel == 0 ? q   : sel == 1 ? kx  : v;
    const __bf16* Wt   = sel == 0 ? WqT : sel == 1 ? WkT : WvT;
    const float*  bias = sel == 0 ? bq  : sel == 1 ? bk  : nullptr;
    __bf16*       outp = sel == 0 ? qh  : sel == 1 ? kh  : vh;

    int wave = threadIdx.x >> 5, lane = threadIdx.x & 31;
    int half = lane >> 4, l16 = lane & 15;
    int row = blockIdx.x * 16 + l16;  // A-fragment row

    const float* ap = X + (size_t)row * kD;

    v8f acc[4] = {{}, {}, {}, {}};
    for (int kc = 0; kc < kD / 32; ++kc) {
        int kb = kc * 32 + half * 16;
        v16bf a;
#pragma unroll
        for (int t = 0; t < 8; ++t) {
            a[t]     = (__bf16)ap[kb + t];
            a[8 + t] = (__bf16)ap[kb + 8 + t];
        }
#pragma unroll
        for (int tn = 0; tn < 4; ++tn) {
            int col = (wave * 4 + tn) * 16 + l16;  // B-fragment column
            v16bf b = *(const v16bf*)(Wt + (size_t)col * kD + kc * 32 + half * 16);
            acc[tn] = WMMA_BF16(a, b, acc[tn]);
        }
    }
#pragma unroll
    for (int tn = 0; tn < 4; ++tn) {
        int col = (wave * 4 + tn) * 16 + l16;
        float bb = bias ? bias[col] : 0.f;
#pragma unroll
        for (int r = 0; r < 8; ++r) {
            int m = blockIdx.x * 16 + r + half * 8;
            outp[(size_t)m * kPROJ + col] = (__bf16)(acc[tn][r] + bb);
        }
    }
}

// ---------------------------------------------------------------------------
// Kernel 2: attention for one (batch*window, head) per block. 4 waves.
// S (f32) and P, V^T (bf16) staged in LDS; softmax in f32.
// ---------------------------------------------------------------------------
__global__ void attention(const __bf16* __restrict__ qh, const __bf16* __restrict__ kh,
                          const __bf16* __restrict__ vh, const unsigned char* __restrict__ mask,
                          __bf16* __restrict__ aout) {
    __shared__ __align__(32) float  S[kLP * kLP];    // 16 KB
    __shared__ __align__(32) __bf16 P[kLP * kLP];    //  8 KB
    __shared__ __align__(32) __bf16 Vt[kHD * kLP];   //  4 KB  Vt[c][k] = V[k][c]

    int bwh = blockIdx.x;
    int bw = bwh >> 3, h = bwh & 7;
    int nw = bw & (kNW - 1);
    const __bf16* qp = qh + (size_t)bw * kL * kPROJ + h * kHD;
    const __bf16* kp = kh + (size_t)bw * kL * kPROJ + h * kHD;
    const __bf16* vp = vh + (size_t)bw * kL * kPROJ + h * kHD;
    __bf16*       op = aout + (size_t)bw * kL * kPROJ + h * kHD;
    const unsigned char* mp = mask + (size_t)nw * kL * kL;

    int tid = threadIdx.x;
    int wave = tid >> 5, lane = tid & 31, half = lane >> 4, l16 = lane & 15;

    // stage V^T into LDS (zero-padded keys 49..63)
    for (int i = tid; i < kHD * kLP; i += 128) {
        int c = i >> 6, kk = i & 63;
        Vt[i] = (kk < kL) ? vp[(size_t)kk * kPROJ + c] : (__bf16)0.f;
    }
    __syncthreads();

    // ---- S = scale * Q K^T  (each wave: 16-row band x 4 col tiles) ----
    int rowA = wave * 16 + l16;
    v16bf a = {};
    if (rowA < kL) {
        const __bf16* ap = qp + (size_t)rowA * kPROJ;
        int kb = half * 16;
#pragma unroll
        for (int t = 0; t < 8; ++t) { a[t] = ap[kb + t]; a[8 + t] = ap[kb + 8 + t]; }
    }
    const float scale = 0.17677669529663687f;  // 1/sqrt(32)
    for (int tn = 0; tn < 4; ++tn) {
        int n = tn * 16 + l16;
        v16bf b = {};
        if (n < kL) b = *(const v16bf*)(kp + (size_t)n * kPROJ + half * 16);
        v8f c = {};
        c = WMMA_BF16(a, b, c);
#pragma unroll
        for (int r = 0; r < 8; ++r) {
            int m = wave * 16 + r + half * 8;
            float s = c[r] * scale;
            if (n >= kL) s = -1e30f;
            else if (m < kL && mp[m * kL + n]) s = -1000.f;
            S[m * kLP + n] = s;
        }
    }
    __syncthreads();

    // ---- softmax: lanes 0-15 of each wave handle one row each ----
    if (half == 0) {
        int m = wave * 16 + l16;
        if (m < kL) {
            float mx = -1e30f;
            for (int n2 = 0; n2 < kLP; ++n2) mx = fmaxf(mx, S[m * kLP + n2]);
            float sum = 0.f;
            for (int n2 = 0; n2 < kLP; ++n2) sum += __expf(S[m * kLP + n2] - mx);
            float inv = 1.f / sum;
            for (int n2 = 0; n2 < kLP; ++n2)
                P[m * kLP + n2] = (__bf16)(__expf(S[m * kLP + n2] - mx) * inv);
        } else {
            for (int n2 = 0; n2 < kLP; ++n2) P[m * kLP + n2] = (__bf16)0.f;
        }
    }
    __syncthreads();

    // ---- O = P @ V  (each wave: its 16-row band, 2 col tiles over HD=32) ----
    for (int tn = 0; tn < 2; ++tn) {
        v8f acc = {};
        for (int kc = 0; kc < 2; ++kc) {
            v16bf pa, vb;
            const __bf16* pp = &P[(wave * 16 + l16) * kLP + kc * 32 + half * 16];
#pragma unroll
            for (int t = 0; t < 8; ++t) { pa[t] = pp[t]; pa[8 + t] = pp[8 + t]; }
            vb = *(const v16bf*)(&Vt[(tn * 16 + l16) * kLP + kc * 32 + half * 16]);
            acc = WMMA_BF16(pa, vb, acc);
        }
#pragma unroll
        for (int r = 0; r < 8; ++r) {
            int m = wave * 16 + r + half * 8;
            if (m < kL) op[(size_t)m * kPROJ + tn * 16 + l16] = (__bf16)acc[r];
        }
    }
}

// ---------------------------------------------------------------------------
// Kernel 3: output projection, fp32 result + bias.  grid=(M/16), block=128.
// Each wave drives one A fragment into 3 N-tiles (12 tiles = 192 cols/block).
// ---------------------------------------------------------------------------
__global__ void out_proj(const __bf16* __restrict__ A, const __bf16* __restrict__ WwT,
                         const float* __restrict__ bw, float* __restrict__ out) {
    int wave = threadIdx.x >> 5, lane = threadIdx.x & 31;
    int half = lane >> 4, l16 = lane & 15;
    int row = blockIdx.x * 16 + l16;

    const __bf16* ap = A + (size_t)row * kPROJ;

    v8f acc[3] = {{}, {}, {}};
    for (int kc = 0; kc < kPROJ / 32; ++kc) {
        v16bf a;
        const __bf16* aq = ap + kc * 32 + half * 16;
#pragma unroll
        for (int t = 0; t < 8; ++t) { a[t] = aq[t]; a[8 + t] = aq[8 + t]; }
#pragma unroll
        for (int tn = 0; tn < 3; ++tn) {
            int col = (wave * 3 + tn) * 16 + l16;
            v16bf b = *(const v16bf*)(WwT + (size_t)col * kPROJ + kc * 32 + half * 16);
            acc[tn] = WMMA_BF16(a, b, acc[tn]);
        }
    }
#pragma unroll
    for (int tn = 0; tn < 3; ++tn) {
        int col = (wave * 3 + tn) * 16 + l16;
        float bb = bw[col];
#pragma unroll
        for (int r = 0; r < 8; ++r) {
            int m = blockIdx.x * 16 + r + half * 8;
            out[(size_t)m * kD + col] = acc[tn][r] + bb;
        }
    }
}

// ---------------------------------------------------------------------------
extern "C" void kernel_launch(void* const* d_in, const int* in_sizes, int n_in,
                              void* d_out, int out_size, void* d_ws, size_t ws_size,
                              hipStream_t stream) {
    (void)in_sizes; (void)n_in; (void)out_size; (void)ws_size;
    const float* query = (const float*)d_in[0];
    const float* key_  = (const float*)d_in[1];
    const float* value = (const float*)d_in[2];
    const unsigned char* mask = (const unsigned char*)d_in[3];
    const float* Wq = (const float*)d_in[4];
    const float* bq = (const float*)d_in[5];
    const float* Wk = (const float*)d_in[6];
    const float* bk = (const float*)d_in[7];
    const float* Wv = (const float*)d_in[8];
    const float* Ww = (const float*)d_in[9];
    const float* bw = (const float*)d_in[10];
    float* out = (float*)d_out;

    char* ws = (char*)d_ws;
    size_t off = 0;
    auto alloc = [&](size_t bytes) {
        char* p = ws + off;
        off += (bytes + 255) & ~(size_t)255;
        return p;
    };
    __bf16* qh  = (__bf16*)alloc((size_t)kM * kPROJ * 2);
    __bf16* kh  = (__bf16*)alloc((size_t)kM * kPROJ * 2);
    __bf16* vh  = (__bf16*)alloc((size_t)kM * kPROJ * 2);
    __bf16* ah  = (__bf16*)alloc((size_t)kM * kPROJ * 2);
    __bf16* WqT = (__bf16*)alloc((size_t)kD * kPROJ * 2);
    __bf16* WkT = (__bf16*)alloc((size_t)kD * kPROJ * 2);
    __bf16* WvT = (__bf16*)alloc((size_t)kD * kPROJ * 2);
    __bf16* WwT = (__bf16*)alloc((size_t)kD * kPROJ * 2);

    pack_weights<<<(4 * kD * kPROJ + 255) / 256, 256, 0, stream>>>(
        Wq, Wk, Wv, Ww, WqT, WkT, WvT, WwT);

    dim3 g1(kM / 16, 1, 3);
    qkv_proj<<<g1, 128, 0, stream>>>(query, key_, value, WqT, WkT, WvT, bq, bk, qh, kh, vh);

    attention<<<kBNW * kHEADS, 128, 0, stream>>>(qh, kh, vh, mask, ah);

    out_proj<<<kM / 16, 128, 0, stream>>>(ah, WwT, bw, out);
}
